// CrossAttention_43585328119966
// MI455X (gfx1250) — matrix operand order repaired and checked by reading, not compiled, not run on hardware
//
#include <hip/hip_runtime.h>

#define DIM 1024
#define HEADS 16
#define DHEAD 64
#define SEQ_N 2048
#define SEQ_M 2048
#define BATCH 2

typedef __attribute__((ext_vector_type(16))) _Float16 v16h;
typedef __attribute__((ext_vector_type(8)))  _Float16 v8h;
typedef __attribute__((ext_vector_type(8)))  float    v8f;

#define WMMA_F16(a, b, c) \
  __builtin_amdgcn_wmma_f32_16x16x32_f16(false, (a), false, (b), (short)0, (c), false, false)

static __device__ __forceinline__ v16h cat8(v8h lo, v8h hi) {
  return __builtin_shufflevector(lo, hi, 0, 1, 2, 3, 4, 5, 6, 7, 8, 9, 10, 11, 12, 13, 14, 15);
}

// A-fragment (16x32 f16, row-major source, leading dim lda in elements).
// Layout: lane l<16 -> row l, K 0..15; lane>=16 -> row l-16, K 16..31 (matches B-frag).
static __device__ __forceinline__ v16h load_a_frag(const _Float16* base, int lda) {
  const int lane = threadIdx.x & 31;
  const _Float16* p = base + (size_t)(lane & 15) * lda + ((lane >> 4) << 4);
  v8h lo = *(const v8h*)p;
  v8h hi = *(const v8h*)(p + 8);
  return cat8(lo, hi);
}

// B-fragment (32x16 f16). Source is column-major-over-K: row n of Bt = column n,
// K contiguous. Lane l<16 -> col l, K 0..15; lane>=16 -> col l-16, K 16..31.
static __device__ __forceinline__ v16h load_b_frag(const _Float16* base, int ldb) {
  const int lane = threadIdx.x & 31;
  return *(const v16h*)(base + (size_t)(lane & 15) * ldb + ((lane >> 4) << 4));
}

// ---------------------------------------------------------------------------
// LayerNorm (one block per row of 1024), f32 in -> f16 out
// ---------------------------------------------------------------------------
__global__ __launch_bounds__(256) void layernorm_f16_kernel(
    const float* __restrict__ X, const float* __restrict__ W,
    const float* __restrict__ Bv, _Float16* __restrict__ out) {
  __shared__ float red[8];
  const size_t row = blockIdx.x;
  const int t = threadIdx.x;
  const float4 v = ((const float4*)(X + row * DIM))[t];

  float s = v.x + v.y + v.z + v.w;
#pragma unroll
  for (int m = 1; m < 32; m <<= 1) s += __shfl_xor(s, m, 32);
  if ((t & 31) == 0) red[t >> 5] = s;
  __syncthreads();
  float tot = 0.f;
#pragma unroll
  for (int i = 0; i < 8; ++i) tot += red[i];
  const float mu = tot * (1.0f / DIM);

  const float dx = v.x - mu, dy = v.y - mu, dz = v.z - mu, dw = v.w - mu;
  float q = dx * dx + dy * dy + dz * dz + dw * dw;
#pragma unroll
  for (int m = 1; m < 32; m <<= 1) q += __shfl_xor(q, m, 32);
  __syncthreads();
  if ((t & 31) == 0) red[t >> 5] = q;
  __syncthreads();
  float vs = 0.f;
#pragma unroll
  for (int i = 0; i < 8; ++i) vs += red[i];
  const float rs = rsqrtf(vs * (1.0f / DIM) + 1e-5f);

  const float4 wv = ((const float4*)W)[t];
  const float4 bv = ((const float4*)Bv)[t];
  _Float16* o = out + row * DIM + (size_t)t * 4;
  o[0] = (_Float16)(dx * rs * wv.x + bv.x);
  o[1] = (_Float16)(dy * rs * wv.y + bv.y);
  o[2] = (_Float16)(dz * rs * wv.z + bv.z);
  o[3] = (_Float16)(dw * rs * wv.w + bv.w);
}

// ---------------------------------------------------------------------------
// Weight convert + transpose: W f32 [K][N]  ->  Wt f16 [N][K]
// ---------------------------------------------------------------------------
__global__ __launch_bounds__(256) void convert_transpose_kernel(
    const float* __restrict__ W, _Float16* __restrict__ Wt, int K, int N) {
  __shared__ float tile[32][33];
  const int n0 = blockIdx.x * 32, k0 = blockIdx.y * 32;
#pragma unroll
  for (int i = threadIdx.y; i < 32; i += 8)
    tile[i][threadIdx.x] = W[(size_t)(k0 + i) * N + n0 + threadIdx.x];
  __syncthreads();
#pragma unroll
  for (int i = threadIdx.y; i < 32; i += 8)
    Wt[(size_t)(n0 + i) * K + k0 + threadIdx.x] = (_Float16)tile[threadIdx.x][i];
}

// ---------------------------------------------------------------------------
// WMMA GEMM: C[M][Nout] = A[M][K](f16,row-major) x Bt[Nout][K](f16)
// One 64x64 tile per wave (4x4 accumulator frags), 4 waves per block.
// MODE 0: f16 out * scale (Q proj). MODE 1: KV split -> K row-major + V
// transposed to [b,h,d,m]. MODE 2: f32 out (final projection).
// ---------------------------------------------------------------------------
template <int MODE>
__global__ __launch_bounds__(128) void gemm64_wmma(
    const _Float16* __restrict__ A, const _Float16* __restrict__ Bt,
    void* __restrict__ O0, void* __restrict__ O1, int K, int Nout, float scale) {
  const int lane = threadIdx.x & 31;
  const int wv = threadIdx.x >> 5;
  const int ln = lane & 15, hi = lane >> 4;
  const int tilesx = Nout >> 6;
  const int tid = blockIdx.x * 4 + wv;
  const size_t row0 = (size_t)(tid / tilesx) << 6;
  const size_t col0 = (size_t)(tid % tilesx) << 6;

  v8f acc[4][4] = {};
  for (int k0 = 0; k0 < K; k0 += 32) {
    v16h a[4], b[4];
#pragma unroll
    for (int rt = 0; rt < 4; ++rt)
      a[rt] = load_a_frag(A + (row0 + rt * 16) * K + k0, K);
#pragma unroll
    for (int ct = 0; ct < 4; ++ct)
      b[ct] = load_b_frag(Bt + (col0 + ct * 16) * K + k0, K);
#pragma unroll
    for (int rt = 0; rt < 4; ++rt)
#pragma unroll
      for (int ct = 0; ct < 4; ++ct)
        acc[rt][ct] = WMMA_F16(a[rt], b[ct], acc[rt][ct]);
  }

#pragma unroll
  for (int rt = 0; rt < 4; ++rt) {
    const size_t rbase = row0 + rt * 16 + hi * 8;  // C layout: lanes>=16 hold rows +8
#pragma unroll
    for (int ct = 0; ct < 4; ++ct) {
      const int c = (int)col0 + ct * 16 + ln;
      if (MODE == 0) {
        _Float16* O = (_Float16*)O0;
#pragma unroll
        for (int v = 0; v < 8; ++v)
          O[(rbase + v) * 1024 + c] = (_Float16)(acc[rt][ct][v] * scale);
      } else if (MODE == 2) {
        float* O = (float*)O0;
#pragma unroll
        for (int v = 0; v < 8; ++v)
          O[(rbase + v) * 1024 + c] = acc[rt][ct][v];
      } else {  // MODE 1: split KV
        if (c < 1024) {
          _Float16* Kh = (_Float16*)O0;
#pragma unroll
          for (int v = 0; v < 8; ++v)
            Kh[(rbase + v) * 1024 + c] = (_Float16)acc[rt][ct][v];
        } else {
          _Float16* Vt = (_Float16*)O1;
          const int da = c - 1024;
          const int hh = da >> 6, dd = da & 63;
#pragma unroll
          for (int v = 0; v < 8; ++v) {
            const size_t r = rbase + v;
            const size_t bb = r >> 11, mr = r & 2047;
            Vt[((bb * HEADS + hh) * DHEAD + dd) * (size_t)SEQ_M + mr] =
                (_Float16)acc[rt][ct][v];
          }
        }
      }
    }
  }
}

// ---------------------------------------------------------------------------
// Flash attention: block = 4 waves = 64 queries; wave owns 16 query rows.
// Q pre-scaled by 1/sqrt(d). Per 32-key tile: S via WMMA, online softmax
// with cross-lane row reductions, P->LDS->A-frag, PV via WMMA.
// ---------------------------------------------------------------------------
__global__ __launch_bounds__(128) void flash_attn_wmma(
    const _Float16* __restrict__ Q, const _Float16* __restrict__ Kh,
    const _Float16* __restrict__ Vt, const unsigned char* __restrict__ mask,
    _Float16* __restrict__ AO) {
  const int lane = threadIdx.x & 31;
  const int wv = threadIdx.x >> 5;
  const int ln = lane & 15, hi = lane >> 4;
  const int bb = blockIdx.z, h = blockIdx.y;
  const int q0 = blockIdx.x * 64 + wv * 16;

  __shared__ _Float16 Plds[4][16 * 32];
  _Float16* pw = &Plds[wv][0];

  const _Float16* Qb = Q + ((size_t)bb * SEQ_N + q0) * DIM + h * DHEAD;
  const v16h aq0 = load_a_frag(Qb, DIM);        // d = 0..31
  const v16h aq1 = load_a_frag(Qb + 32, DIM);   // d = 32..63

  const _Float16* Kb = Kh + ((size_t)bb * SEQ_M) * DIM + h * DHEAD;
  const _Float16* Vb = Vt + ((size_t)(bb * HEADS + h) * DHEAD) * SEQ_M;
  const unsigned char* mb = mask + (size_t)bb * SEQ_M;

  float mst[8], lst[8];
  v8f o[4] = {};
#pragma unroll
  for (int v = 0; v < 8; ++v) { mst[v] = -1e30f; lst[v] = 0.f; }

  for (int j0 = 0; j0 < SEQ_M; j0 += 32) {
    v8f s[2];
#pragma unroll
    for (int jt = 0; jt < 2; ++jt) {
      // B-frag of K^T: column j holds contiguous d values (K row-major).
      const _Float16* kp = Kb + (size_t)(j0 + jt * 16 + ln) * DIM + hi * 16;
      v16h b0 = *(const v16h*)kp;         // d 0..31 chunk
      v16h b1 = *(const v16h*)(kp + 32);  // d 32..63 chunk
      v8f z = {};
      s[jt] = WMMA_F16(aq0, b0, z);
      s[jt] = WMMA_F16(aq1, b1, s[jt]);
      if (!mb[j0 + jt * 16 + ln]) {       // this lane's whole column is one key
#pragma unroll
        for (int v = 0; v < 8; ++v) s[jt][v] = -1e30f;
      }
    }

    // Row max across the 16 lanes that share each row (xor masks stay in-half).
    float pm[8];
#pragma unroll
    for (int v = 0; v < 8; ++v) pm[v] = fmaxf(s[0][v], s[1][v]);
#pragma unroll
    for (int v = 0; v < 8; ++v)
#pragma unroll
      for (int m = 1; m <= 8; m <<= 1)
        pm[v] = fmaxf(pm[v], __shfl_xor(pm[v], m, 32));

    float corr[8];
#pragma unroll
    for (int v = 0; v < 8; ++v) {
      const float nm = fmaxf(mst[v], pm[v]);
      corr[v] = __expf(mst[v] - nm);
      mst[v] = nm;
    }

    v8f p0, p1;
    float ts[8];
#pragma unroll
    for (int v = 0; v < 8; ++v) {
      p0[v] = __expf(s[0][v] - mst[v]);
      p1[v] = __expf(s[1][v] - mst[v]);
      ts[v] = p0[v] + p1[v];
    }
#pragma unroll
    for (int v = 0; v < 8; ++v)
#pragma unroll
      for (int m = 1; m <= 8; m <<= 1) ts[v] += __shfl_xor(ts[v], m, 32);
#pragma unroll
    for (int v = 0; v < 8; ++v) lst[v] = lst[v] * corr[v] + ts[v];
#pragma unroll
    for (int nt = 0; nt < 4; ++nt)
#pragma unroll
      for (int v = 0; v < 8; ++v) o[nt][v] *= corr[v];

    // C-layout P -> LDS (row = hi*8+v, col = ln / 16+ln), reload as A-frag.
    // LDS ops from the same wave complete in order; tile is wave-private.
#pragma unroll
    for (int v = 0; v < 8; ++v) {
      pw[(hi * 8 + v) * 32 + ln] = (_Float16)p0[v];
      pw[(hi * 8 + v) * 32 + 16 + ln] = (_Float16)p1[v];
    }
    const v16h ap = load_a_frag(pw, 32);

#pragma unroll
    for (int nt = 0; nt < 4; ++nt) {
      // B-frag of V: column n = output d; contiguous j thanks to Vt layout.
      const v16h bv = *(const v16h*)(Vb + (size_t)(nt * 16 + ln) * SEQ_M + j0 + hi * 16);
      o[nt] = WMMA_F16(ap, bv, o[nt]);
    }
  }

  _Float16* Ob = AO + ((size_t)bb * SEQ_N + q0) * DIM + h * DHEAD;
#pragma unroll
  for (int v = 0; v < 8; ++v) {
    const float inv = 1.0f / lst[v];
#pragma unroll
    for (int nt = 0; nt < 4; ++nt)
      Ob[(size_t)(hi * 8 + v) * DIM + nt * 16 + ln] = (_Float16)(o[nt][v] * inv);
  }
}

// ---------------------------------------------------------------------------
extern "C" void kernel_launch(void* const* d_in, const int* in_sizes, int n_in,
                              void* d_out, int out_size, void* d_ws, size_t ws_size,
                              hipStream_t stream) {
  const float* x = (const float*)d_in[0];
  const float* context = (const float*)d_in[1];
  const float* norm_w = (const float*)d_in[2];
  const float* norm_b = (const float*)d_in[3];
  const float* ctx_norm_w = (const float*)d_in[4];
  const float* ctx_norm_b = (const float*)d_in[5];
  const float* Wq = (const float*)d_in[6];
  const float* Wkv = (const float*)d_in[7];
  const float* Wo = (const float*)d_in[8];
  const unsigned char* cmask = (const unsigned char*)d_in[9];

  // Workspace layout (f16), total 56 MB.
  _Float16* ws = (_Float16*)d_ws;
  _Float16* xn = ws;                                  // [4096][1024]
  _Float16* cn = xn + (size_t)4096 * 1024;            // [4096][1024]
  _Float16* Wq_t = cn + (size_t)4096 * 1024;          // [1024][1024]
  _Float16* Wkv_t = Wq_t + (size_t)1024 * 1024;       // [2048][1024]
  _Float16* Wo_t = Wkv_t + (size_t)2048 * 1024;       // [1024][1024]
  _Float16* Qh = Wo_t + (size_t)1024 * 1024;          // [4096][1024] (pre-scaled)
  _Float16* Kh = Qh + (size_t)4096 * 1024;            // [4096][1024]
  _Float16* Vt = Kh + (size_t)4096 * 1024;            // [b][h][d][m]
  _Float16* AO = Vt + (size_t)4096 * 1024;            // [4096][1024]

  layernorm_f16_kernel<<<4096, 256, 0, stream>>>(x, norm_w, norm_b, xn);
  layernorm_f16_kernel<<<4096, 256, 0, stream>>>(context, ctx_norm_w, ctx_norm_b, cn);

  convert_transpose_kernel<<<dim3(32, 32), dim3(32, 8), 0, stream>>>(Wq, Wq_t, 1024, 1024);
  convert_transpose_kernel<<<dim3(64, 32), dim3(32, 8), 0, stream>>>(Wkv, Wkv_t, 1024, 2048);
  convert_transpose_kernel<<<dim3(32, 32), dim3(32, 8), 0, stream>>>(Wo, Wo_t, 1024, 1024);

  const float scale = 0.125f;  // 64^-0.5, folded into Q
  gemm64_wmma<0><<<(64 * 16) / 4, 128, 0, stream>>>(xn, Wq_t, (void*)Qh, nullptr, 1024, 1024, scale);
  gemm64_wmma<1><<<(64 * 32) / 4, 128, 0, stream>>>(cn, Wkv_t, (void*)Kh, (void*)Vt, 1024, 2048, 1.0f);

  flash_attn_wmma<<<dim3(SEQ_N / 64, HEADS, BATCH), 128, 0, stream>>>(Qh, Kh, Vt, cmask, AO);

  gemm64_wmma<2><<<(64 * 16) / 4, 128, 0, stream>>>(AO, Wo_t, d_out, nullptr, 1024, 1024, 1.0f);

  (void)in_sizes; (void)n_in; (void)out_size; (void)ws_size;
}
